// MyModel_70265664963044
// MI455X (gfx1250) — matrix operand order repaired and checked by reading, not compiled, not run on hardware
//
#include <hip/hip_runtime.h>
#include <math.h>

#define NN 512
#define NEDGE (512 * 511)

typedef __attribute__((ext_vector_type(16))) _Float16 v16h;
typedef __attribute__((ext_vector_type(8)))  float    v8f;

// Branchless tanh: 1 - 2/(exp(2x)+1).  exp overflow -> +1, underflow -> -1 (no NaN).
__device__ __forceinline__ float fast_tanh(float x) {
    float e = __expf(2.f * x);
    return 1.f - 2.f * __builtin_amdgcn_rcpf(e + 1.f);
}

// ---------------------------------------------------------------------------
// Kernel 0: build eta/phi arrays and inp_node x0 = [node_feat[:,1:14], emb[flav]]
// ---------------------------------------------------------------------------
__global__ void prep_kernel(const float* __restrict__ nf, const float* __restrict__ emb,
                            float* __restrict__ eta, float* __restrict__ phi,
                            float* __restrict__ x0) {
    int n = blockIdx.x * blockDim.x + threadIdx.x;
    if (n >= NN) return;
    const float* row = nf + n * 14;
    const float FL[4] = {5.f, 4.f, 0.f, 15.f};
    float fl = row[0];
    int idx = 0;
    for (int t = 3; t >= 1; --t) if (fl == FL[t]) idx = t; // first-match (values unique)
    eta[n] = row[2];
    phi[n] = row[3];
    for (int k = 0; k < 13; ++k) x0[n * 16 + k] = row[1 + k];
    for (int k = 0; k < 3; ++k)  x0[n * 16 + 13 + k] = emb[idx * 3 + k];
}

// ---------------------------------------------------------------------------
// Kernel 1: per-edge dr (shared by all three layers)
// ---------------------------------------------------------------------------
__global__ void dr_kernel(const float* __restrict__ eta, const float* __restrict__ phi,
                          float* __restrict__ dr) {
    int e = blockIdx.x * blockDim.x + threadIdx.x;
    if (e >= NEDGE) return;
    int i = e / 511, jp = e - i * 511;
    int j = jp + (jp >= i ? 1 : 0);
    float de = eta[i] - eta[j];
    float dp = phi[i] - phi[j];
    const float PIv = 3.14159265358979323846f;
    if (dp >= PIv) dp -= 2.f * PIv;
    if (dp < -PIv) dp += 2.f * PIv;
    dr[e] = sqrtf(de * de + dp * dp);
}

// ---------------------------------------------------------------------------
// Kernel 2: P = X@W1[0:Fin] + b1 ; Q = X@W1[Fin:2Fin]   (tiny GEMMs)
// ---------------------------------------------------------------------------
__global__ __launch_bounds__(128) void pq_kernel(const float* __restrict__ xin, int Fin,
                                                 const float* __restrict__ W1,
                                                 const float* __restrict__ b1, int emid,
                                                 float* __restrict__ P, float* __restrict__ Q) {
    int n = blockIdx.x, t = threadIdx.x;
    __shared__ float sx[80];
    for (int k = t; k < Fin; k += 128) sx[k] = xin[n * Fin + k];
    __syncthreads();
    if (t < emid) {
        float p = b1[t], q = 0.f;
        for (int k = 0; k < Fin; ++k) {
            p += sx[k] * W1[k * emid + t];
            q += sx[k] * W1[(Fin + k) * emid + t];
        }
        P[n * emid + t] = p;
        Q[n * emid + t] = q;
    }
}

// ---------------------------------------------------------------------------
// Kernel 3 (hot): per node i, msg[i] = sum_j tanh( relu(P_i+Q_j+dr*w1c) @ W2 + b2 )
// Templated on EMID/KF so all fragment arrays are statically indexed registers.
// Padded edge rows carry sQ = -inf so relu() zeroes their A rows branchlessly.
// ---------------------------------------------------------------------------
template <int EMID, int KF>
__global__ __launch_bounds__(128) void edge_msg_kernel(
    const float* __restrict__ P, const float* __restrict__ Q,
    const float* __restrict__ dr, const float* __restrict__ w1c,
    const float* __restrict__ W2, const float* __restrict__ b2,
    float* __restrict__ msg) {
    const int i = blockIdx.x;
    const int tid = threadIdx.x;
    const int wave = tid >> 5;
    const int lane = tid & 31;

    __shared__ _Float16 sW2[EMID * 32];                 // e_w2 as f16, row-major [k][n]
    __shared__ __align__(16) float sP[EMID];
    __shared__ __align__(16) float sC[EMID];
    __shared__ float sB2[32];
    __shared__ __align__(16) float sQ[4][16 * EMID];    // per-wave Q tile
    __shared__ float sMsg[32];

    for (int t = tid; t < EMID; t += 128) {
        sP[t] = P[i * EMID + t];
        sC[t] = w1c[t];
    }
    for (int t = tid; t < EMID * 32; t += 128) sW2[t] = (_Float16)W2[t];
    if (tid < 32) { sB2[tid] = b2[tid]; sMsg[tid] = 0.f; }
    __syncthreads();

    // B fragments (constant across all tiles): contiguous k order, same as A:
    // lane L holds column N = L%16; half h -> K = (L<16?0:16)+h within each K-32 block.
    v16h bf[2][KF];
#pragma unroll
    for (int nt = 0; nt < 2; ++nt)
#pragma unroll
        for (int kf = 0; kf < KF; ++kf) {
            v16h bv;
            int ncol = nt * 16 + (lane & 15);
            int kb = kf * 32 + ((lane < 16) ? 0 : 16);
#pragma unroll
            for (int h = 0; h < 16; ++h) bv[h] = sW2[(kb + h) * 32 + ncol];
            bf[nt][kf] = bv;
        }

    const int M = lane & 15;
    const float b2c0 = sB2[lane & 15];
    const float b2c1 = sB2[16 + (lane & 15)];
    float macc0 = 0.f, macc1 = 0.f;      // per-lane column sums, N = {lane%16, 16+lane%16}

    constexpr int Q4 = EMID / 4;         // float4 per Q row
    const float4* __restrict__ Qv = reinterpret_cast<const float4*>(Q);
    float4* sQv = reinterpret_cast<float4*>(&sQ[wave][0]);
    const float NEG_INF = -__builtin_inff();

    for (int jt = wave; jt < 32; jt += 4) {
        // prefetch next tile's Q rows into cache (global_prefetch_b8)
        {
            int jtn = jt + 4;
            int jp2 = jtn * 16 + (lane & 15);
            if (jtn < 32 && lane < 16 && jp2 < 511) {
                int j2 = jp2 + (jp2 >= i ? 1 : 0);
                __builtin_prefetch(&Qv[j2 * Q4], 0, 3);
            }
        }
        // stage 16 Q rows (b128 global loads -> b128 LDS stores), wave-private slab.
        // Rows beyond 510 get -inf so the A build zeroes them without a select.
#pragma unroll
        for (int t = lane; t < 16 * Q4; t += 32) {
            int m = t / Q4, c4 = t - m * Q4;
            int jp = jt * 16 + m;
            int jpc = jp < 511 ? jp : 510;           // clamp (branchless)
            int j = jpc + (jpc >= i ? 1 : 0);
            float4 q = Qv[j * Q4 + c4];
            if (jp >= 511) q = make_float4(NEG_INF, NEG_INF, NEG_INF, NEG_INF);
            sQv[t] = q;
        }
        // wave-local: LDS pipeline is in-order per wave; wait for stores, block reordering
        asm volatile("s_wait_dscnt 0x0" ::: "memory");

        int jpM = jt * 16 + M;
        float drv = dr[(size_t)i * 511 + (jpM < 511 ? jpM : 510)];  // clamped; -inf absorbs

        // A fragments: contiguous k order, same as B: lane L -> row M = L%16;
        // half h -> K = (L<16?0:16) + h within each K-32 block.
        v16h af[KF];
#pragma unroll
        for (int kf = 0; kf < KF; ++kf) {
            v16h av;
            int kb = kf * 32 + ((lane < 16) ? 0 : 16);
#pragma unroll
            for (int h = 0; h < 16; ++h) {
                int c = kb + h;
                float v = sP[c] + sQ[wave][M * EMID + c] + drv * sC[c];
                v = fmaxf(v, 0.f);                   // relu; fmax(-inf,0)=0 pads rows
                av[h] = (_Float16)v;
            }
            af[kf] = av;
        }

        v8f acc0 = {0.f, 0.f, 0.f, 0.f, 0.f, 0.f, 0.f, 0.f};
        v8f acc1 = {0.f, 0.f, 0.f, 0.f, 0.f, 0.f, 0.f, 0.f};
#pragma unroll
        for (int kf = 0; kf < KF; ++kf) {
            acc0 = __builtin_amdgcn_wmma_f32_16x16x32_f16(false, af[kf], false, bf[0][kf],
                                                          (short)0, acc0, false, false);
            acc1 = __builtin_amdgcn_wmma_f32_16x16x32_f16(false, af[kf], false, bf[1][kf],
                                                          (short)0, acc1, false, false);
        }

        // Epilogue: D layout lane L -> N = L%16, vgpr r -> M = r + (L<16?0:8)
        int mbase = (lane < 16) ? 0 : 8;
#pragma unroll
        for (int r = 0; r < 8; ++r) {
            int jp = jt * 16 + mbase + r;
            float t0 = fast_tanh(acc0[r] + b2c0);
            float t1 = fast_tanh(acc1[r] + b2c1);
            macc0 += (jp < 511) ? t0 : 0.f;          // cndmask, no branch
            macc1 += (jp < 511) ? t1 : 0.f;
        }
    }

    // lanes L and L+16 hold the same output column -> fold, then LDS atomic across waves
    macc0 += __shfl_xor(macc0, 16, 32);
    macc1 += __shfl_xor(macc1, 16, 32);
    if (lane < 16) {
        atomicAdd(&sMsg[lane], macc0);
        atomicAdd(&sMsg[16 + lane], macc1);
    }
    __syncthreads();
    if (tid < 32) msg[i * 32 + tid] = sMsg[tid];
}

// ---------------------------------------------------------------------------
// Kernel 4: node MLPs + normalization + concat:  x_next = [x0, normalize([out1,out2])]
// ---------------------------------------------------------------------------
__global__ __launch_bounds__(64) void node_update_kernel(
    const float* __restrict__ xin, int Fin, const float* __restrict__ x0,
    const float* __restrict__ msg,
    const float* __restrict__ n1w1, const float* __restrict__ n1b1,
    const float* __restrict__ n1w2, const float* __restrict__ n1b2, int nmid,
    const float* __restrict__ n2w1, const float* __restrict__ n2b1,
    const float* __restrict__ n2w2, const float* __restrict__ n2b2,
    float* __restrict__ xout) {
    int n = blockIdx.x, t = threadIdx.x;
    __shared__ float sx[80], sm[32], sh1[56], sh2[32], so[64], snorm;
    for (int k = t; k < Fin; k += 64) sx[k] = xin[n * Fin + k];
    if (t < 32) sm[t] = msg[n * 32 + t];
    __syncthreads();
    if (t < nmid) {
        float a = n1b1[t];
        for (int k = 0; k < Fin; ++k) a += sx[k] * n1w1[k * nmid + t];
        sh1[t] = a > 0.f ? a : 0.f;
    }
    if (t < 32) {
        float a = n2b1[t];
        for (int k = 0; k < 32; ++k) a += sm[k] * n2w1[k * 32 + t];
        sh2[t] = a > 0.f ? a : 0.f;
    }
    __syncthreads();
    if (t < 32) {
        float a = n1b2[t];
        for (int k = 0; k < nmid; ++k) a += sh1[k] * n1w2[k * 32 + t];
        so[t] = tanhf(a);
    } else {
        int u = t - 32;
        float a = n2b2[u];
        for (int k = 0; k < 32; ++k) a += sh2[k] * n2w2[k * 32 + u];
        so[t] = tanhf(a);
    }
    __syncthreads();
    if (t == 0) {
        float s = 0.f;
        for (int k = 0; k < 64; ++k) s += so[k] * so[k];
        snorm = 1.f / sqrtf(s);
    }
    __syncthreads();
    for (int k = t; k < 80; k += 64)
        xout[n * 80 + k] = (k < 16) ? x0[n * 16 + k] : so[k - 16] * snorm;
}

// ---------------------------------------------------------------------------
// Kernel 5: correlator MLP 80 -> 128 -> 64 -> 1, sigmoid
// ---------------------------------------------------------------------------
__global__ __launch_bounds__(128) void corr_kernel(
    const float* __restrict__ x,
    const float* __restrict__ w0, const float* __restrict__ b0,
    const float* __restrict__ w1, const float* __restrict__ b1,
    const float* __restrict__ w2, const float* __restrict__ b2,
    float* __restrict__ out) {
    int n = blockIdx.x, t = threadIdx.x;
    __shared__ float sx[80], sh1[128], sh2[64];
    if (t < 80) sx[t] = x[n * 80 + t];
    __syncthreads();
    {
        float a = b0[t];
        for (int k = 0; k < 80; ++k) a += sx[k] * w0[k * 128 + t];
        sh1[t] = a > 0.f ? a : 0.f;
    }
    __syncthreads();
    if (t < 64) {
        float a = b1[t];
        for (int k = 0; k < 128; ++k) a += sh1[k] * w1[k * 64 + t];
        sh2[t] = a > 0.f ? a : 0.f;
    }
    __syncthreads();
    if (t == 0) {
        float a = b2[0];
        for (int k = 0; k < 64; ++k) a += sh2[k] * w2[k];
        out[n] = 1.f / (1.f + expf(-a));
    }
}

// ---------------------------------------------------------------------------
// Host launcher: resolve input flattening order by size-matching, run pipeline
// ---------------------------------------------------------------------------
static const int FinA[3]  = {16, 80, 80};
static const int emidA[3] = {32, 96, 96};
static const int nmidA[3] = {24, 56, 56};

static int slotSize(int s) {
    if (s == 0) return NN * 14;
    if (s == 1) return 12;
    if (s >= 38) {
        static const int cs[6] = {80 * 128, 128, 128 * 64, 64, 64, 1};
        return cs[s - 38];
    }
    int l = (s - 2) / 12, k = (s - 2) % 12;
    int F = FinA[l], E = emidA[l], Nm = nmidA[l];
    switch (k) {
        case 0: return (2 * F + 1) * E;  case 1: return E;
        case 2: return E * 32;           case 3: return 32;
        case 4: return F * Nm;           case 5: return Nm;
        case 6: return Nm * 32;          case 7: return 32;
        case 8: return 32 * 32;          case 9: return 32;
        case 10: return 32 * 32;         case 11: return 32;
    }
    return 0;
}

extern "C" void kernel_launch(void* const* d_in, const int* in_sizes, int n_in,
                              void* d_out, int out_size, void* d_ws, size_t ws_size,
                              hipStream_t stream) {
    (void)out_size; (void)ws_size;

    // candidate flatten orders (slot ids):
    int orderA[44], orderB[44];
    {
        int a = 0;
        orderA[a++] = 0; orderA[a++] = 1;                       // node_feat, emb
        for (int l = 0; l < 3; ++l)
            for (int k = 0; k < 12; ++k) orderA[a++] = 2 + l * 12 + k;  // insertion order
        for (int k = 0; k < 6; ++k) orderA[a++] = 38 + k;       // corr
    }
    {
        int b = 0;
        orderB[b++] = 0;                                        // node_feat
        for (int k = 0; k < 6; ++k) orderB[b++] = 38 + k;       // corr (sorted: c<e<g)
        orderB[b++] = 1;                                        // emb
        static const int perm[12] = {1, 3, 0, 2, 5, 7, 4, 6, 9, 11, 8, 10}; // sorted keys
        for (int l = 0; l < 3; ++l)
            for (int k = 0; k < 12; ++k) orderB[b++] = 2 + l * 12 + perm[k];
    }
    const int* order = orderA;
    if (n_in == 44) {
        bool okA = true, okB = true;
        for (int p = 0; p < 44; ++p) {
            if (in_sizes[p] != slotSize(orderA[p])) okA = false;
            if (in_sizes[p] != slotSize(orderB[p])) okB = false;
        }
        order = okA ? orderA : (okB ? orderB : orderA);
    }
    const float* ptr[44];
    for (int p = 0; p < 44 && p < n_in; ++p) ptr[order[p]] = (const float*)d_in[p];

    // workspace layout (floats)
    float* w   = (float*)d_ws;
    float* eta = w;                 // 512
    float* phi = eta + NN;          // 512
    float* x0  = phi + NN;          // 512*16
    float* x   = x0 + NN * 16;      // 512*80
    float* drb = x + NN * 80;       // 512*511
    float* P   = drb + NEDGE;       // 512*96
    float* Q   = P + NN * 96;       // 512*96
    float* msg = Q + NN * 96;       // 512*32

    prep_kernel<<<2, 256, 0, stream>>>(ptr[0], ptr[1], eta, phi, x0);
    dr_kernel<<<(NEDGE + 255) / 256, 256, 0, stream>>>(eta, phi, drb);

    const float* xin = x0;
    for (int l = 0; l < 3; ++l) {
        const int base = 2 + l * 12;
        const float* ew1  = ptr[base + 0];
        const float* eb1  = ptr[base + 1];
        const float* ew2  = ptr[base + 2];
        const float* eb2  = ptr[base + 3];
        const float* n1w1 = ptr[base + 4];
        const float* n1b1 = ptr[base + 5];
        const float* n1w2 = ptr[base + 6];
        const float* n1b2 = ptr[base + 7];
        const float* n2w1 = ptr[base + 8];
        const float* n2b1 = ptr[base + 9];
        const float* n2w2 = ptr[base + 10];
        const float* n2b2 = ptr[base + 11];
        int F = FinA[l], E = emidA[l], Nm = nmidA[l];
        const float* w1c = ew1 + (size_t)(2 * F) * E;   // last row of e_w1 (dr weight)

        pq_kernel<<<NN, 128, 0, stream>>>(xin, F, ew1, eb1, E, P, Q);
        if (E == 32)
            edge_msg_kernel<32, 1><<<NN, 128, 0, stream>>>(P, Q, drb, w1c, ew2, eb2, msg);
        else
            edge_msg_kernel<96, 3><<<NN, 128, 0, stream>>>(P, Q, drb, w1c, ew2, eb2, msg);
        node_update_kernel<<<NN, 64, 0, stream>>>(xin, F, x0, msg,
                                                  n1w1, n1b1, n1w2, n1b2, Nm,
                                                  n2w1, n2b1, n2w2, n2b2, x);
        xin = x;
    }

    corr_kernel<<<NN, 128, 0, stream>>>(x, ptr[38], ptr[39], ptr[40], ptr[41],
                                        ptr[42], ptr[43], (float*)d_out);
}
